// S2VT_37288906064356
// MI455X (gfx1250) — matrix operand order repaired and checked by reading, not compiled, not run on hardware
//
#include <hip/hip_runtime.h>

// ---------------------------------------------------------------------------
// S2VT (video captioning): 2-layer LSTM + projections, CDNA5 (gfx1250) WMMA.
//
// Precision: weights/activations rounded f32->bf16 for the matrix engine,
// all accumulation in f32 via v_wmma_f32_16x16x32_bf16. Gates/state in f32.
//
// Phases:
//   (1) big parallel WMMA GEMMs for input projections (xg1, xg2, logits);
//       64x64 output per wave (4x4 wmma tiles) -> 16 b128 loads : 16 wmma
//   (2) persistent cooperative kernel for the 109-step recurrences:
//       per step:  G = h_prev @ W_hh^T  (WMMA, 512 waves / 1024 tiles)
//                  grid barrier -> fused gate elementwise -> grid barrier
// ---------------------------------------------------------------------------

typedef __bf16 bf16;
typedef __attribute__((ext_vector_type(8)))  __bf16 v8bf;
typedef __attribute__((ext_vector_type(16))) __bf16 v16bf;
typedef __attribute__((ext_vector_type(8)))  float  v8f;

#define NB  128    // batch
#define TVV 80     // video frames
#define FF  1280   // feature dim
#define HH  512    // hidden
#define GG  2048   // 4*H gates
#define TT  109    // total timesteps (80 + 29)
#define LCC 29     // caption steps (CAPLEN-1)
#define VV  1500   // vocab

// ---------------- WMMA fragment helpers (CDNA5 wave32 layouts) -------------
__device__ __forceinline__ v16bf cat8(v8bf lo, v8bf hi) {
  return __builtin_shufflevector(lo, hi, 0,1,2,3,4,5,6,7,8,9,10,11,12,13,14,15);
}

// A: 16x32 bf16, row-major in memory (lda elements).
// lane<16: M=lane, K = {k0..k0+7} U {k0+16..k0+23}; lane>=16: +8 on K runs.
__device__ __forceinline__ v16bf load_a_frag(const bf16* __restrict__ A, long lda,
                                             int m0, int k0, int lane) {
  int half = lane >> 4;
  const bf16* p = A + (long)(m0 + (lane & 15)) * lda + k0 + 8 * half;
  return cat8(*(const v8bf*)p, *(const v8bf*)(p + 16));
}

// B: 32x16 bf16 (KxN) with weights stored N-major ([n][k], ldb elements).
// lane's 16 values = W[n][k0+8*half .. +8) U W[n][k0+16+8*half .. +8)  -> two b128 loads.
__device__ __forceinline__ v16bf load_b_frag(const bf16* __restrict__ W, long ldb,
                                             int n0, int k0, int lane, int nclamp) {
  int col = n0 + (lane & 15);
  col = col < nclamp ? col : nclamp - 1;
  const bf16* p = W + (long)col * ldb + k0 + 8 * (lane >> 4);
  return cat8(*(const v8bf*)p, *(const v8bf*)(p + 16));
}

__device__ __forceinline__ v8f wmma_bf16(v16bf a, v16bf b, v8f c) {
  return __builtin_amdgcn_wmma_f32_16x16x32_bf16(false, a, false, b,
                                                 (short)0, c, false, false);
}

// ---------------- activation helpers ---------------------------------------
__device__ __forceinline__ float fsig(float x)  { return 1.f / (1.f + __expf(-x)); }
__device__ __forceinline__ float ftanh(float x) { float e = __expf(2.f * x); return 1.f - 2.f / (e + 1.f); }

// ---------------- device-wide barrier (persistent recurrence kernel) -------
__device__ __forceinline__ void grid_barrier(unsigned* cnt, unsigned* gen, unsigned nb) {
  __threadfence();
  __syncthreads();
  if (threadIdx.x == 0) {
    unsigned g = __hip_atomic_load(gen, __ATOMIC_RELAXED, __HIP_MEMORY_SCOPE_AGENT);
    unsigned a = __hip_atomic_fetch_add(cnt, 1u, __ATOMIC_ACQ_REL, __HIP_MEMORY_SCOPE_AGENT);
    if (a == nb - 1u) {
      __hip_atomic_store(cnt, 0u, __ATOMIC_RELAXED, __HIP_MEMORY_SCOPE_AGENT);
      __hip_atomic_store(gen, g + 1u, __ATOMIC_RELEASE, __HIP_MEMORY_SCOPE_AGENT);
    } else {
      unsigned cur;
      do {
        __builtin_amdgcn_s_sleep(2);
        cur = __hip_atomic_load(gen, __ATOMIC_ACQUIRE, __HIP_MEMORY_SCOPE_AGENT);
      } while (cur == g);
    }
  }
  __syncthreads();
}

// ---------------- generic WMMA GEMM: C[m,n] = A[m,:]·B[:,n] (+bias) --------
// One wave computes a 64(M) x 64(N) block: 4x4 wmma tiles, K in steps of 32.
// All call sites have M % 64 == 0.
// MODE 0: Out[m*ldc+n] = acc + bias[n]
// MODE 1: xg1 rows   : b=m/80, t=m%80 -> Out[(b*109+t)*ldc+n] = acc + bias[n]
// MODE 2: xg2 tail   : b=m/29, j=m%29 -> Out[(b*109+80+j)*ldc+n] += acc
// MODE 3: logits     : b=m/29, t=m%29 -> Out[(b*1500+n)*29+t] = acc + bias[n]  (n<Nreal)
template<int MODE>
__global__ void __launch_bounds__(256)
gemm_wmma(const bf16* __restrict__ A, long lda,
          const bf16* __restrict__ Bw, long ldb,
          const float* __restrict__ bias,
          float* __restrict__ Out, long ldc,
          int M, int Nreal, int K, int NT4) {
  int wave = (int)((blockIdx.x * blockDim.x + threadIdx.x) >> 5);
  int lane = threadIdx.x & 31;
  int mt4 = wave / NT4;
  int n4  = wave % NT4;
  if (mt4 * 64 >= M) return;
  int m0 = mt4 * 64, n0 = n4 * 64;
  int half = lane >> 4, nc = lane & 15;

  v8f acc[4][4];
  if (MODE == 2) {
    for (int mi = 0; mi < 4; ++mi)
      for (int ni = 0; ni < 4; ++ni)
        for (int r = 0; r < 8; ++r) {
          int m = m0 + mi * 16 + r + 8 * half;
          int n = n0 + ni * 16 + nc;
          int b = m / LCC, j = m % LCC;
          acc[mi][ni][r] = Out[((long)(b * TT + TVV + j)) * ldc + n];
        }
  } else {
    for (int ni = 0; ni < 4; ++ni) {
      int n = n0 + ni * 16 + nc;
      float bv = (bias && n < Nreal) ? bias[n] : 0.f;
      for (int mi = 0; mi < 4; ++mi)
        for (int r = 0; r < 8; ++r) acc[mi][ni][r] = bv;
    }
  }

  for (int k0 = 0; k0 < K; k0 += 32) {
    v16bf af[4], bfv[4];
    for (int mi = 0; mi < 4; ++mi) af[mi] = load_a_frag(A, lda, m0 + mi * 16, k0, lane);
    for (int ni = 0; ni < 4; ++ni) bfv[ni] = load_b_frag(Bw, ldb, n0 + ni * 16, k0, lane, Nreal);
    for (int mi = 0; mi < 4; ++mi)
      for (int ni = 0; ni < 4; ++ni)
        acc[mi][ni] = wmma_bf16(af[mi], bfv[ni], acc[mi][ni]);
  }

  for (int mi = 0; mi < 4; ++mi)
    for (int ni = 0; ni < 4; ++ni)
      for (int r = 0; r < 8; ++r) {
        int m = m0 + mi * 16 + r + 8 * half;
        int n = n0 + ni * 16 + nc;
        float v = acc[mi][ni][r];
        if (MODE == 0) {
          Out[(long)m * ldc + n] = v;
        } else if (MODE == 1) {
          int b = m / TVV, t = m % TVV;
          Out[((long)(b * TT + t)) * ldc + n] = v;
        } else if (MODE == 2) {
          int b = m / LCC, j = m % LCC;
          Out[((long)(b * TT + TVV + j)) * ldc + n] = v;
        } else { // MODE 3
          if (n < Nreal) {
            int b = m / LCC, t = m % LCC;
            Out[((long)(b * VV + n)) * LCC + t] = v;
          }
        }
      }
}

// ---------------- persistent LSTM recurrence (one layer) -------------------
// Grid must be exactly 64 blocks x 256 threads (512 waves == 1024 16x16 tiles
// of the [128 x 2048] gate matrix, 2 tiles/wave).
__global__ void __launch_bounds__(256)
lstm_seq(const float* __restrict__ xg,       // [NB][TT][GG] preactivations (incl. biases)
         const bf16* __restrict__ WhhN,      // [GG][HH] bf16, N-major
         bf16* __restrict__ hbuf,            // [NB][HH]
         float* __restrict__ cbuf,           // [NB][HH]
         float* __restrict__ Gbuf,           // [NB][GG]
         bf16* __restrict__ outbuf,          // layer1: [NB][TT][HH]; layer2: [NB][LCC][HH]
         int outT, int tOutStart,
         unsigned* __restrict__ syncCnt, unsigned* __restrict__ syncGen) {
  const int tid  = threadIdx.x;
  const int wid  = (blockIdx.x << 3) + (tid >> 5);   // 0..511
  const int lane = tid & 31;
  const int half = lane >> 4, nc = lane & 15;
  const int m0 = (wid >> 6) * 16;                    // batch tile base (0..112)
  const int n0 = (wid & 63) * 32;                    // gate-col base (0..2016)
  const int nthreads = gridDim.x * blockDim.x;       // 16384

  for (int t = 0; t < TT; ++t) {
    // ---- G = h_prev @ Whh^T (skip at t==0: h_prev == 0) ----
    v8f acc0 = {}, acc1 = {};
    if (t > 0) {
      for (int k0 = 0; k0 < HH; k0 += 32) {
        v16bf a  = load_a_frag(hbuf, HH, m0, k0, lane);
        v16bf b0 = load_b_frag(WhhN, HH, n0,      k0, lane, GG);
        v16bf b1 = load_b_frag(WhhN, HH, n0 + 16, k0, lane, GG);
        acc0 = wmma_bf16(a, b0, acc0);
        acc1 = wmma_bf16(a, b1, acc1);
      }
    }
    for (int r = 0; r < 8; ++r) {
      int m = m0 + r + 8 * half;
      Gbuf[(long)m * GG + n0 + nc]      = acc0[r];
      Gbuf[(long)m * GG + n0 + 16 + nc] = acc1[r];
    }
    grid_barrier(syncCnt, syncGen, gridDim.x);

    // ---- fused gates: i,f,g,o -> c,h ----
    for (int idx = (int)(blockIdx.x * blockDim.x) + tid; idx < NB * HH; idx += nthreads) {
      int b = idx >> 9, h = idx & (HH - 1);
      long gb = (long)b * GG;
      long xb = ((long)b * TT + t) * GG;
      float gi = Gbuf[gb + h]           + xg[xb + h];
      float gf = Gbuf[gb + 512 + h]     + xg[xb + 512 + h];
      float gg = Gbuf[gb + 1024 + h]    + xg[xb + 1024 + h];
      float go = Gbuf[gb + 1536 + h]    + xg[xb + 1536 + h];
      float cprev = (t == 0) ? 0.f : cbuf[idx];
      float c  = fsig(gf) * cprev + fsig(gi) * ftanh(gg);
      float hn = fsig(go) * ftanh(c);
      cbuf[idx] = c;
      bf16 hb = (bf16)hn;
      hbuf[idx] = hb;
      if (t >= tOutStart)
        outbuf[((long)b * outT + (t - tOutStart)) * HH + h] = hb;
    }
    grid_barrier(syncCnt, syncGen, gridDim.x);
  }
}

// ---------------- small elementwise kernels ---------------------------------
__global__ void f32_to_bf16_k(const float* __restrict__ s, bf16* __restrict__ d, long n) {
  long stride = (long)gridDim.x * blockDim.x;
  for (long i = (long)blockIdx.x * blockDim.x + threadIdx.x; i < n; i += stride)
    d[i] = (bf16)s[i];
}

__global__ void embed_gather_k(const int* __restrict__ cap, const float* __restrict__ emb,
                               bf16* __restrict__ out) {
  long n = (long)NB * LCC * HH;
  long stride = (long)gridDim.x * blockDim.x;
  for (long i = (long)blockIdx.x * blockDim.x + threadIdx.x; i < n; i += stride) {
    int h = (int)(i & (HH - 1));
    int r = (int)(i >> 9);           // b*LCC + j
    int b = r / LCC, j = r % LCC;
    int tok = cap[b * (LCC + 1) + j];            // caption[:, :-1]
    out[i] = (bf16)emb[(long)tok * HH + h];
  }
}

__global__ void combine_bias_k(const float* __restrict__ a, const float* __restrict__ b,
                               float* __restrict__ o, int n) {
  int i = blockIdx.x * blockDim.x + threadIdx.x;
  if (i < n) o[i] = a[i] + b[i];
}

// Fill xg1 rows t in [80,109) with bias1 (zero video input there) + reset sync.
__global__ void init_tail_sync_k(float* __restrict__ xg, const float* __restrict__ bias1,
                                 unsigned* __restrict__ sync) {
  long n = (long)NB * LCC * GG;
  long stride = (long)gridDim.x * blockDim.x;
  long tid0 = (long)blockIdx.x * blockDim.x + threadIdx.x;
  for (long i = tid0; i < n; i += stride) {
    int g = (int)(i & (GG - 1));
    int r = (int)(i >> 11);          // b*LCC + j
    int b = r / LCC, j = r % LCC;
    xg[((long)(b * TT + TVV + j)) * GG + g] = bias1[g];
  }
  if (tid0 < 8) sync[tid0] = 0u;
}

// ---------------- host side --------------------------------------------------
extern "C" void kernel_launch(void* const* d_in, const int* in_sizes, int n_in,
                              void* d_out, int out_size, void* d_ws, size_t ws_size,
                              hipStream_t stream) {
  (void)in_sizes; (void)n_in; (void)out_size; (void)ws_size;

  const float* vf   = (const float*)d_in[0];
  const int*   cap  = (const int*)  d_in[1];
  const float* Wih1 = (const float*)d_in[2];
  const float* Whh1 = (const float*)d_in[3];
  const float* bih1 = (const float*)d_in[4];
  const float* bhh1 = (const float*)d_in[5];
  const float* Wih2 = (const float*)d_in[6];
  const float* Whh2 = (const float*)d_in[7];
  const float* bih2 = (const float*)d_in[8];
  const float* bhh2 = (const float*)d_in[9];
  const float* emb  = (const float*)d_in[10];
  const float* Wlin = (const float*)d_in[11];
  const float* blin = (const float*)d_in[12];
  float* out = (float*)d_out;

  // workspace carve-up (~180 MB total)
  char* ws = (char*)d_ws;
  size_t off = 0;
  auto carve = [&](size_t bytes) { void* p = ws + off; off += (bytes + 255) & ~(size_t)255; return p; };
  float* xg      = (float*)carve((size_t)NB * TT * GG * 4);     // xg1 then reused as xg2
  bf16*  wih1b   = (bf16*) carve((size_t)GG * FF * 2);
  bf16*  whh1b   = (bf16*) carve((size_t)GG * HH * 2);
  bf16*  wih2b   = (bf16*) carve((size_t)GG * 2 * HH * 2);
  bf16*  whh2b   = (bf16*) carve((size_t)GG * HH * 2);
  bf16*  wlinb   = (bf16*) carve((size_t)VV * HH * 2);
  bf16*  vinb    = (bf16*) carve((size_t)NB * TVV * FF * 2);
  bf16*  vout    = (bf16*) carve((size_t)NB * TT * HH * 2);
  bf16*  embcap  = (bf16*) carve((size_t)NB * LCC * HH * 2);
  bf16*  capout  = (bf16*) carve((size_t)NB * LCC * HH * 2);
  float* bias1   = (float*)carve((size_t)GG * 4);
  float* bias2   = (float*)carve((size_t)GG * 4);
  bf16*  hbuf    = (bf16*) carve((size_t)NB * HH * 2);
  float* cbuf    = (float*)carve((size_t)NB * HH * 4);
  float* Gbuf    = (float*)carve((size_t)NB * GG * 4);
  unsigned* sync = (unsigned*)carve(256);
  unsigned* syncCnt = sync, *syncGen = sync + 1;

  const int EW = 256;
  // weight / activation conversion to bf16 (layouts already N-major / row-major)
  f32_to_bf16_k<<<2048, EW, 0, stream>>>(Wih1, wih1b, (long)GG * FF);
  f32_to_bf16_k<<<1024, EW, 0, stream>>>(Whh1, whh1b, (long)GG * HH);
  f32_to_bf16_k<<<1024, EW, 0, stream>>>(Wih2, wih2b, (long)GG * 2 * HH);
  f32_to_bf16_k<<<1024, EW, 0, stream>>>(Whh2, whh2b, (long)GG * HH);
  f32_to_bf16_k<<<1024, EW, 0, stream>>>(Wlin, wlinb, (long)VV * HH);
  f32_to_bf16_k<<<2048, EW, 0, stream>>>(vf,   vinb,  (long)NB * TVV * FF);
  embed_gather_k<<<1024, EW, 0, stream>>>(cap, emb, embcap);
  combine_bias_k<<<8, EW, 0, stream>>>(bih1, bhh1, bias1, GG);
  combine_bias_k<<<8, EW, 0, stream>>>(bih2, bhh2, bias2, GG);
  init_tail_sync_k<<<1024, EW, 0, stream>>>(xg, bias1, sync);

  // xg1 = vin @ W_ih1^T + bias1   (M=10240, N=2048, K=1280): 160*32 waves
  gemm_wmma<1><<<640, 256, 0, stream>>>(vinb, FF, wih1b, FF, bias1, xg, GG,
                                        NB * TVV, GG, FF, 32);
  // LSTM layer 1 -> vout (bf16 [NB][TT][HH])
  lstm_seq<<<64, 256, 0, stream>>>(xg, whh1b, hbuf, cbuf, Gbuf, vout, TT, 0,
                                   syncCnt, syncGen);
  // xg2 (reuses xg buffer): top half from video_out, all t (M=13952): 218*32 waves
  gemm_wmma<0><<<872, 256, 0, stream>>>(vout, HH, wih2b, 2 * HH, bias2, xg, GG,
                                        NB * TT, GG, HH, 32);
  // ... += emb_cap @ W_ih2[:,512:]^T for t>=80 (M=3712): 58*32 waves
  gemm_wmma<2><<<232, 256, 0, stream>>>(embcap, HH, wih2b + HH, 2 * HH, nullptr, xg, GG,
                                        NB * LCC, GG, HH, 32);
  // LSTM layer 2 -> capout (bf16 [NB][LCC][HH]), outputs only t>=80
  lstm_seq<<<64, 256, 0, stream>>>(xg, whh2b, hbuf, cbuf, Gbuf, capout, LCC, TVV,
                                   syncCnt, syncGen);
  // logits: out[b][v][t] = capout[b][t] · W_lin[v] + b_lin[v] (M=3712, N=1500): 58*24 waves
  gemm_wmma<3><<<174, 256, 0, stream>>>(capout, HH, wlinb, HH, blin, out, 0,
                                        NB * LCC, VV, HH, 24);
}
